// CapsuleLayer_44719199486642
// MI455X (gfx1250) — matrix operand order repaired and checked by reading, not compiled, not run on hardware
//
#include <hip/hip_runtime.h>
#include <hip/hip_bf16.h>
#include <math.h>

// ---------------------------------------------------------------------------
// CapsuleLayer fused kernel for gfx1250 (MI455X).
//  priors GEMM via v_wmma_f32_16x16x32_bf16 with bf16 hi/lo split (near-fp32),
//  dynamic routing kept in registers per wave (wave32, one capsule per wave).
// ---------------------------------------------------------------------------

typedef __attribute__((ext_vector_type(16))) __bf16 bf16x16;
typedef __attribute__((ext_vector_type(8)))  float  floatx8;

#define C_IN   128
#define NC     512          // NCAPS*CCH
#define NCAPS_ 32
#define CCH_   16
#define HOUT   32
#define WOUT   32
#define BATCH  8
#define HP     34           // padded H
#define WP     34           // padded W
#define NIT_   3

// workspace element counts / byte offsets
#define XSZ  (BATCH*HP*WP*C_IN)        // 1,183,744 bf16 elements per buffer
#define WSZ  (9*NCAPS_*4*32*16)        // 589,824 bf16 elements per buffer
#define XHI_OFF  ((size_t)0)
#define XLO_OFF  ((size_t)XSZ*2)
#define WHI_OFF  ((size_t)XSZ*4)
#define WLO_OFF  (WHI_OFF + (size_t)WSZ*2)

// --- prep: NCHW f32 -> padded NHWC bf16 hi/lo -------------------------------
__global__ void caps_prep_x(const float* __restrict__ in,
                            __bf16* __restrict__ xhi, __bf16* __restrict__ xlo) {
    int idx = blockIdx.x * 256 + threadIdx.x;       // exact grid, no tail
    int c = idx & (C_IN - 1);
    int s = idx >> 7;
    int x = s % WP;
    int t = s / WP;
    int y = t % HP;
    int b = t / HP;
    float v = 0.0f;
    if (x >= 1 && x <= WOUT && y >= 1 && y <= HOUT)
        v = in[((b * C_IN + c) * HOUT + (y - 1)) * WOUT + (x - 1)];
    __bf16 h = (__bf16)v;
    __bf16 l = (__bf16)(v - (float)h);
    xhi[idx] = h;
    xlo[idx] = l;
}

// --- prep: weights [3,3,128,512] f32 -> per-lane A-fragment order, bf16 hi/lo
// A fragment (16-bit A 16x32, contiguous k order, same as B): lane L, element e (0..15):
//   K = (L>=16 ? 16 : 0) + e,  M = L&15
// flat index: ((((tap*32+cap)*4 + kc)*32 + lane)*16 + e)
__global__ void caps_prep_w(const float* __restrict__ w,
                            __bf16* __restrict__ whi, __bf16* __restrict__ wlo) {
    int idx  = blockIdx.x * 256 + threadIdx.x;      // exact grid
    int e    = idx & 15;
    int lane = (idx >> 4) & 31;
    int kc   = (idx >> 9) & 3;
    int cap  = (idx >> 11) & 31;
    int tap  = idx >> 16;                           // 0..8
    int kl   = ((lane >> 4) << 4) + e;
    int c    = kc * 32 + kl;
    int n    = cap * CCH_ + (lane & 15);
    float v  = w[(tap * C_IN + c) * NC + n];
    __bf16 h = (__bf16)v;
    __bf16 l = (__bf16)(v - (float)h);
    whi[idx] = h;
    wlo[idx] = l;
}

// --- main: priors (WMMA) + bias + 3 routing iterations + store --------------
// grid: 2048 blocks x 256 threads. block -> (tile = bid>>2, capgroup = bid&3)
// tile = b*64 + h*2 + wt  (16 consecutive w positions).  wave -> one capsule.
__global__ void __launch_bounds__(256)
caps_main(const __bf16* __restrict__ xhi, const __bf16* __restrict__ xlo,
          const __bf16* __restrict__ whi, const __bf16* __restrict__ wlo,
          const float* __restrict__ bias, float* __restrict__ out) {
    const int lane  = threadIdx.x & 31;
    const int wave  = threadIdx.x >> 5;
    const int bid   = blockIdx.x;
    const int tile  = bid >> 2;                 // 0..511
    const int cap   = (bid & 3) * 8 + wave;     // 0..31
    const int wt    = tile & 1;
    const int h     = (tile >> 1) & 31;
    const int b     = tile >> 6;
    const int w0    = wt * 16;
    const int pos   = lane & 15;                // N (spatial position in tile)
    const int khalf = lane >> 4;                // which half of K / of M

    const floatx8 z = {0.f, 0.f, 0.f, 0.f, 0.f, 0.f, 0.f, 0.f};
    floatx8 acc[9];
#pragma unroll
    for (int t = 0; t < 9; ++t) acc[t] = z;

#pragma unroll
    for (int t = 0; t < 9; ++t) {
        const int i = t / 3, j = t % 3;
        // B fragment: fixed position (w0+j+pos, h+i), 16 consecutive channels
        const int xoff = ((b * HP + (h + i)) * WP + (w0 + j + pos)) * C_IN + khalf * 16;
        const __bf16* xh = xhi + xoff;
        const __bf16* xl = xlo + xoff;
        // A fragment: pre-swizzled, contiguous per lane
        const int aoff = (((t * NCAPS_ + cap) * 4) * 32 + lane) * 16;
        const __bf16* ah = whi + aoff;
        const __bf16* al = wlo + aoff;

        if (t < 8) {  // prefetch next tap's activation row (global_prefetch_b8)
            const int tn = t + 1;
            __builtin_prefetch(
                xhi + ((b * HP + (h + tn / 3)) * WP + (w0 + tn % 3 + pos)) * C_IN, 0, 1);
        }

#pragma unroll
        for (int kc = 0; kc < 4; ++kc) {
            bf16x16 Ah = *(const bf16x16*)(ah + kc * (32 * 16));
            bf16x16 Al = *(const bf16x16*)(al + kc * (32 * 16));
            bf16x16 Bh = *(const bf16x16*)(xh + kc * 32);
            bf16x16 Bl = *(const bf16x16*)(xl + kc * 32);
            // bf16x3 split accumulation in f32: hi*hi + hi*lo + lo*hi
            acc[t] = __builtin_amdgcn_wmma_f32_16x16x32_bf16(
                false, Ah, false, Bh, (short)0, acc[t], false, false);
            acc[t] = __builtin_amdgcn_wmma_f32_16x16x32_bf16(
                false, Ah, false, Bl, (short)0, acc[t], false, false);
            acc[t] = __builtin_amdgcn_wmma_f32_16x16x32_bf16(
                false, Al, false, Bh, (short)0, acc[t], false, false);
        }
        // priors += bias  (C-tile row M=r+8*khalf -> channel within capsule)
#pragma unroll
        for (int r = 0; r < 8; ++r)
            acc[t][r] += bias[t * NC + cap * CCH_ + khalf * 8 + r];
    }

    // ---- dynamic routing, fully in registers --------------------------------
    floatx8 o = z;
#pragma unroll
    for (int t = 0; t < 9; ++t)
#pragma unroll
        for (int r = 0; r < 8; ++r) o[r] += acc[t][r];
#pragma unroll
    for (int r = 0; r < 8; ++r) o[r] *= (1.0f / 9.0f);

    for (int it = 0; it < NIT_; ++it) {
        float cw[9];
        float csum = 0.0f;
#pragma unroll
        for (int t = 0; t < 9; ++t) {
            float ss = 0.0f;
#pragma unroll
            for (int r = 0; r < 8; ++r) {
                float d = o[r] - acc[t][r];
                ss += d * d;
            }
            // lanes 0-15 hold channels 0-7, lanes 16-31 channels 8-15:
            // combine halves so both carry the full 16-channel sum.
            ss += __shfl_xor(ss, 16, 32);
            cw[t] = 1.0f / sqrtf(ss + 1e-4f);
            csum += cw[t];
        }
        float inv = 1.0f / csum;
        floatx8 no = z;
#pragma unroll
        for (int t = 0; t < 9; ++t) {
            float c = cw[t] * inv;
#pragma unroll
            for (int r = 0; r < 8; ++r) no[r] += acc[t][r] * c;
        }
        o = no;
    }

    // ---- store [B, NC, H, W] ------------------------------------------------
#pragma unroll
    for (int r = 0; r < 8; ++r) {
        int ch = cap * CCH_ + khalf * 8 + r;
        out[((b * NC + ch) * HOUT + h) * WOUT + w0 + pos] = o[r];
    }
}

extern "C" void kernel_launch(void* const* d_in, const int* in_sizes, int n_in,
                              void* d_out, int out_size, void* d_ws, size_t ws_size,
                              hipStream_t stream) {
    (void)in_sizes; (void)n_in; (void)out_size; (void)ws_size;
    const float* input  = (const float*)d_in[0];   // [8,128,32,32]
    const float* weight = (const float*)d_in[1];   // [1,1,1,3,3,128,512]
    const float* bias   = (const float*)d_in[2];   // [1,1,1,3,3,1,512]
    float* out = (float*)d_out;                    // [8,512,32,32]
    char* ws = (char*)d_ws;

    __bf16* xhi = (__bf16*)(ws + XHI_OFF);
    __bf16* xlo = (__bf16*)(ws + XLO_OFF);
    __bf16* whi = (__bf16*)(ws + WHI_OFF);
    __bf16* wlo = (__bf16*)(ws + WLO_OFF);

    caps_prep_x<<<XSZ / 256, 256, 0, stream>>>(input, xhi, xlo);
    caps_prep_w<<<WSZ / 256, 256, 0, stream>>>(weight, whi, wlo);
    caps_main<<<512 * 4, 256, 0, stream>>>(xhi, xlo, whi, wlo, bias, out);
}
